// H3HybridTransform_24730421690754
// MI455X (gfx1250) — matrix operand order repaired and checked by reading, not compiled, not run on hardware
//
#include <hip/hip_runtime.h>
#include <hip/hip_fp16.h>

typedef __attribute__((ext_vector_type(16))) _Float16 v16h;
typedef __attribute__((ext_vector_type(8)))  _Float16 v8h;
typedef __attribute__((ext_vector_type(8)))  float    v8f;

union V16H { v16h v; v8h p[2]; _Float16 h[16]; };
union V8F  { v8f  v; float f[8]; };

#define BATCH 12288
#define DPAD  96      // 92 padded: K loop = 3 x 32; row stride 192 B (16B aligned)

__device__ __forceinline__ float gelu_exact(float v) {
    return 0.5f * v * (1.0f + erff(v * 0.70710678118654752f));
}

// ---------------------------------------------------------------------------
// Kernel 0: pad + convert one layer's weights to f16 [128][128] (zero-filled)
// and biases to f32 [128] (zero-filled). Removes ALL predication from the
// MLP hot loop: B fragments become unconditional global_load_b128 pairs.
// ---------------------------------------------------------------------------
__global__ __launch_bounds__(256)
void pad_weights_kernel(const float* __restrict__ cw, const float* __restrict__ cb,
                        const float* __restrict__ sw, const float* __restrict__ sb,
                        int din, int dout,
                        _Float16* __restrict__ cwp, _Float16* __restrict__ swp,
                        float* __restrict__ cbp, float* __restrict__ sbp)
{
    const int idx = blockIdx.x * 256 + threadIdx.x;   // 128*128 grid-stride not needed
    if (idx < 128 * 128) {
        const int n = idx >> 7, k = idx & 127;
        cwp[idx] = (n < dout && k < din) ? (_Float16)cw[n * din + k] : (_Float16)0.f;
        swp[idx] = (n < dout)            ? (_Float16)sw[n * 128 + k] : (_Float16)0.f;
    }
    if (idx < 128) {
        cbp[idx] = (idx < dout) ? cb[idx] : 0.f;
        sbp[idx] = (idx < dout) ? sb[idx] : 0.f;
    }
}

// ---------------------------------------------------------------------------
// Kernel 1: fused 4-layer MLP (Linear -> exact GELU + 0.1 * input-skip GEMM)
// One block = 16 batch rows, 128 threads (4 waves), WMMA f16 K=128 zero-pad.
// All fragment loads are unconditional 16B vector loads (padded operands).
// ---------------------------------------------------------------------------
__global__ __launch_bounds__(128)
void mlp_kernel(const float* __restrict__ x,
                const _Float16* __restrict__ cwp,  // 4 x [128][128] f16
                const _Float16* __restrict__ swp,  // 4 x [128][128] f16
                const float* __restrict__ cbp,     // 4 x [128] f32
                const float* __restrict__ sbp,     // 4 x [128] f32
                float* __restrict__ comp_f32,
                _Float16* __restrict__ comp_f16,
                float* __restrict__ norms)
{
    __shared__ _Float16 bufA[16][128];     // f16 x tile (skip-path A operand)
    __shared__ _Float16 bufs[2][16][128];  // ping-pong activations

    const int tid  = threadIdx.x;
    const int lane = tid & 31;
    const int wid  = tid >> 5;
    const int lg   = lane >> 4;   // lane group 0/1
    const int ll   = lane & 15;
    const int i0   = blockIdx.x * 16;

    for (int e = tid; e < 16 * 128; e += 128) {
        int r = e >> 7, c = e & 127;
        _Float16 hv = (_Float16)x[(size_t)(i0 + r) * 128 + c];
        bufA[r][c]    = hv;
        bufs[0][r][c] = hv;
    }
    __syncthreads();

    int curb = 0;
    #pragma unroll
    for (int s = 0; s < 4; ++s) {
        const _Float16* cwL = cwp + (size_t)s * 128 * 128;
        const _Float16* swL = swp + (size_t)s * 128 * 128;
        const float*    cbL = cbp + s * 128;
        const float*    sbL = sbp + s * 128;

        // each wave covers 2 of the 8 N-tiles (128 output cols, zero padded)
        for (int nt = wid; nt < 8; nt += 4) {
            const int n = nt * 16 + ll;   // this lane's output column

            V8F accM, accS;
            #pragma unroll
            for (int r = 0; r < 8; ++r) { accM.f[r] = 0.f; accS.f[r] = 0.f; }

            const _Float16* cwRow = cwL + (size_t)n * 128;
            const _Float16* swRow = swL + (size_t)n * 128;

            #pragma unroll
            for (int t = 0; t < 4; ++t) {          // K = 128 in 4 steps of 32
                const int kb = 32 * t;
                V16H a, a2, bm, bs;
                // A fragments (CDNA5 16-bit A layout): two aligned 16B LDS runs
                a.p[0]  = *(const v8h*)&bufs[curb][ll][kb + 8 * lg];
                a.p[1]  = *(const v8h*)&bufs[curb][ll][kb + 16 + 8 * lg];
                a2.p[0] = *(const v8h*)&bufA[ll][kb + 8 * lg];
                a2.p[1] = *(const v8h*)&bufA[ll][kb + 16 + 8 * lg];
                // B fragments: same K layout as A, two 8-wide runs per lane (zero-padded)
                bm.p[0] = *(const v8h*)(cwRow + kb + 8 * lg);
                bm.p[1] = *(const v8h*)(cwRow + kb + 16 + 8 * lg);
                bs.p[0] = *(const v8h*)(swRow + kb + 8 * lg);
                bs.p[1] = *(const v8h*)(swRow + kb + 16 + 8 * lg);

                accM.v = __builtin_amdgcn_wmma_f32_16x16x32_f16(
                    false, a.v,  false, bm.v, (short)0, accM.v, false, false);
                accS.v = __builtin_amdgcn_wmma_f32_16x16x32_f16(
                    false, a2.v, false, bs.v, (short)0, accS.v, false, false);
            }

            // padded columns: acc==0 and bias==0 -> gelu(0)+0.1*0 == 0, no guard
            const float bias  = cbL[n];
            const float sbias = sbL[n];
            #pragma unroll
            for (int r = 0; r < 8; ++r) {
                const int m = r + 8 * lg;   // C layout: lanes 16-31 hold M=r+8
                float o = gelu_exact(accM.f[r] + bias) + 0.1f * (accS.f[r] + sbias);
                bufs[curb ^ 1][m][n] = (_Float16)o;
            }
        }
        __syncthreads();
        curb ^= 1;
    }

    // emit compressed (padded to 96 cols, cols 92..95 are zero) + norms
    for (int e = tid; e < 16 * DPAD; e += 128) {
        int r = e / DPAD, c = e % DPAD;
        float v = (float)bufs[curb][r][c];
        comp_f32[(size_t)(i0 + r) * DPAD + c] = v;
        comp_f16[(size_t)(i0 + r) * DPAD + c] = (_Float16)v;
    }
    if (tid < 16) {
        float s = 0.f;
        for (int c = 0; c < 92; ++c) {
            float v = (float)bufs[curb][tid][c];
            s = fmaf(v, v, s);
        }
        norms[i0 + tid] = s;
    }
}

// ---------------------------------------------------------------------------
// Kernel 2: basin mixture (tiny): distances to 8 centers, softmax, energy
// ---------------------------------------------------------------------------
__global__ __launch_bounds__(256)
void basin_kernel(const float* __restrict__ comp,
                  const float* __restrict__ basin_c,
                  const float* __restrict__ basin_d,
                  const float* __restrict__ basin_w,
                  float* __restrict__ out_energy,
                  float* __restrict__ out_probs)
{
    const int i = blockIdx.x * 256 + threadIdx.x;
    if (i >= BATCH) return;
    float a[8];
    float amax = -1e30f;
    #pragma unroll
    for (int b = 0; b < 8; ++b) {
        float d2 = 0.f;
        for (int c = 0; c < 92; ++c) {
            float df = comp[(size_t)i * DPAD + c] - basin_c[b * 92 + c];
            d2 = fmaf(df, df, d2);
        }
        float d = sqrtf(fmaxf(d2, 0.f));
        float w = log1pf(expf(basin_w[b]));   // softplus
        a[b] = -d / w;
        amax = fmaxf(amax, a[b]);
    }
    float se = 0.f;
    #pragma unroll
    for (int b = 0; b < 8; ++b) { a[b] = expf(a[b] - amax); se += a[b]; }
    const float inv = 1.f / se;
    float en = 0.f;
    #pragma unroll
    for (int b = 0; b < 8; ++b) {
        float p = a[b] * inv;
        out_probs[(size_t)i * 8 + b] = p;
        en = fmaf(p, basin_d[b], en);
    }
    out_energy[i] = en;
}

// ---------------------------------------------------------------------------
// Kernel 3: fused self-Gram + shell histogram + feature epilogue.
// Block = 16 rows (A frags in registers), 8 waves sweep disjoint 16-col tiles.
// d^2 = ni + nj - 2*dot (dot via WMMA f16, K=96). Histogram on d^2 vs bnd^2.
// comp f16 is 2.25 MB -> resident in 192 MB L2; no B x B matrix materialized.
// ---------------------------------------------------------------------------
__global__ __launch_bounds__(256)
void coord_kernel(const _Float16* __restrict__ compH,
                  const float* __restrict__ compF,
                  const float* __restrict__ norms,
                  const float* __restrict__ shell_bnd,
                  const float* __restrict__ se_w,
                  const float* __restrict__ se_b,
                  float* __restrict__ out_feats,
                  float* __restrict__ out_occ)
{
    __shared__ float occ_l[16][3];

    const int tid  = threadIdx.x;
    const int lane = tid & 31;
    const int wid  = tid >> 5;
    const int lg   = lane >> 4;
    const int ll   = lane & 15;
    const int i0   = blockIdx.x * 16;

    if (tid < 48) occ_l[tid / 3][tid % 3] = 0.f;
    __syncthreads();

    float b2[4];
    #pragma unroll
    for (int k = 0; k < 4; ++k) {
        float bb = log1pf(expf(shell_bnd[k]));  // softplus boundary
        b2[k] = bb * bb;
    }

    // A fragments for this block's 16 rows: 3 K-steps of 32 over K=96
    V16H afr[3];
    {
        const _Float16* rp = compH + (size_t)(i0 + ll) * DPAD;
        #pragma unroll
        for (int t = 0; t < 3; ++t) {
            afr[t].p[0] = *(const v8h*)(rp + 32 * t + 8 * lg);
            afr[t].p[1] = *(const v8h*)(rp + 32 * t + 16 + 8 * lg);
        }
    }
    float ni[8];
    #pragma unroll
    for (int r = 0; r < 8; ++r) ni[r] = norms[i0 + r + 8 * lg];

    float cnt[8][3];
    #pragma unroll
    for (int r = 0; r < 8; ++r)
        #pragma unroll
        for (int s = 0; s < 3; ++s) cnt[r][s] = 0.f;

    for (int jt = wid; jt < BATCH / 16; jt += 8) {
        const int j0 = jt * 16;
        const _Float16* bp = compH + (size_t)(j0 + ll) * DPAD;
        const float nj = norms[j0 + ll];

        if (jt + 8 < BATCH / 16)   // prefetch this wave's next column tile
            __builtin_prefetch(compH + (size_t)(j0 + 128 + ll) * DPAD, 0, 1);

        V8F acc;
        #pragma unroll
        for (int r = 0; r < 8; ++r) acc.f[r] = 0.f;

        #pragma unroll
        for (int t = 0; t < 3; ++t) {
            V16H bfr;   // same K layout as A: two 8-wide runs per lane, 16B-aligned
            bfr.p[0] = *(const v8h*)(bp + 32 * t + 8 * lg);
            bfr.p[1] = *(const v8h*)(bp + 32 * t + 16 + 8 * lg);
            acc.v = __builtin_amdgcn_wmma_f32_16x16x32_f16(
                false, afr[t].v, false, bfr.v, (short)0, acc.v, false, false);
        }

        #pragma unroll
        for (int r = 0; r < 8; ++r) {
            const float d2 = fmaxf(ni[r] + nj - 2.f * acc.f[r], 0.f);
            #pragma unroll
            for (int s = 0; s < 3; ++s)
                cnt[r][s] += (d2 >= b2[s] && d2 < b2[s + 1]) ? 1.f : 0.f;
        }
    }

    #pragma unroll
    for (int r = 0; r < 8; ++r) {
        const int m = r + 8 * lg;
        #pragma unroll
        for (int s = 0; s < 3; ++s)
            atomicAdd(&occ_l[m][s], cnt[r][s]);   // ds_add_f32
    }
    __syncthreads();

    const float scale = 1.0f / (1.081f * (float)(BATCH - 1) + 1e-8f);
    if (tid < 48) {
        int m = tid / 3, s = tid % 3;
        out_occ[(size_t)(i0 + m) * 3 + s] = occ_l[m][s] * scale;
    }

    for (int e = tid; e < 16 * 92; e += 256) {
        const int m = e / 92, c = e % 92;
        const float o0 = occ_l[m][0] * scale;
        const float o1 = occ_l[m][1] * scale;
        const float o2 = occ_l[m][2] * scale;
        float f = compF[(size_t)(i0 + m) * DPAD + c]
                + o0 * se_w[c * 3 + 0] + o1 * se_w[c * 3 + 1] + o2 * se_w[c * 3 + 2]
                + se_b[c];
        out_feats[(size_t)(i0 + m) * 92 + c] = f;
    }
}

// ---------------------------------------------------------------------------
extern "C" void kernel_launch(void* const* d_in, const int* in_sizes, int n_in,
                              void* d_out, int out_size, void* d_ws, size_t ws_size,
                              hipStream_t stream)
{
    const float* x   = (const float*)d_in[0];
    const float* CW[4] = {(const float*)d_in[1],  (const float*)d_in[5],
                          (const float*)d_in[9],  (const float*)d_in[13]};
    const float* CB[4] = {(const float*)d_in[2],  (const float*)d_in[6],
                          (const float*)d_in[10], (const float*)d_in[14]};
    const float* SW[4] = {(const float*)d_in[3],  (const float*)d_in[7],
                          (const float*)d_in[11], (const float*)d_in[15]};
    const float* SB[4] = {(const float*)d_in[4],  (const float*)d_in[8],
                          (const float*)d_in[12], (const float*)d_in[16]};
    const float* shell_bnd = (const float*)d_in[17];
    const float* se_w = (const float*)d_in[18];
    const float* se_b = (const float*)d_in[19];
    const float* basin_c = (const float*)d_in[20];
    const float* basin_d = (const float*)d_in[21];
    const float* basin_w = (const float*)d_in[22];

    float* out        = (float*)d_out;
    float* out_feats  = out;                                 // [B,92]
    float* out_energy = out + (size_t)BATCH * 92;            // [B]
    float* out_probs  = out + (size_t)BATCH * 93;            // [B,8]
    float* out_occ    = out + (size_t)BATCH * 101;           // [B,3]

    // workspace layout (16B-aligned sections)
    char* ws = (char*)d_ws;
    size_t off = 0;
    float*    comp_f32 = (float*)(ws + off);    off += (size_t)BATCH * DPAD * 4;  // 4.72 MB
    float*    norms    = (float*)(ws + off);    off += (size_t)BATCH * 4;         // 48 KB
    _Float16* comp_f16 = (_Float16*)(ws + off); off += (size_t)BATCH * DPAD * 2;  // 2.36 MB
    _Float16* cwp      = (_Float16*)(ws + off); off += (size_t)4 * 128 * 128 * 2; // 128 KB
    _Float16* swp      = (_Float16*)(ws + off); off += (size_t)4 * 128 * 128 * 2; // 128 KB
    float*    cbp      = (float*)(ws + off);    off += (size_t)4 * 128 * 4;
    float*    sbp      = (float*)(ws + off);    off += (size_t)4 * 128 * 4;

    const int DIN[4]  = {128, 118, 109, 100};
    const int DOUT[4] = {118, 109, 100, 92};
    for (int s = 0; s < 4; ++s) {
        pad_weights_kernel<<<(128 * 128 + 255) / 256, 256, 0, stream>>>(
            CW[s], CB[s], SW[s], SB[s], DIN[s], DOUT[s],
            cwp + (size_t)s * 128 * 128, swp + (size_t)s * 128 * 128,
            cbp + s * 128, sbp + s * 128);
    }

    mlp_kernel<<<BATCH / 16, 128, 0, stream>>>(
        x, cwp, swp, cbp, sbp, comp_f32, comp_f16, norms);

    basin_kernel<<<BATCH / 256, 256, 0, stream>>>(
        comp_f32, basin_c, basin_d, basin_w, out_energy, out_probs);

    coord_kernel<<<BATCH / 16, 256, 0, stream>>>(
        comp_f16, comp_f32, norms, shell_bnd, se_w, se_b, out_feats, out_occ);
}
